// HealpixTransformer_25864293056831
// MI455X (gfx1250) — matrix operand order repaired and checked by reading, not compiled, not run on hardware
//
#include <hip/hip_runtime.h>
#include <hip/hip_bf16.h>
#include <math.h>

// ---------------------------------------------------------------------------
// MI455X (gfx1250) implementation.
// Pipeline: sample+cross  ->  GEMM-A (ifft axis -2 via WMMA)  ->
//           GEMM-B (irfft axis -1 via WMMA)  ->  per-batch top-4.
// Matrix work: v_wmma_f32_16x16x32_f16 (wave32, 16x16 tiles, K=32).
// B-side operands are stored pre-swizzled in the exact WMMA B-fragment
// layout so each fragment load is one aligned 32-byte vector load.
// ---------------------------------------------------------------------------

typedef __attribute__((ext_vector_type(16))) _Float16 v16h;
typedef __attribute__((ext_vector_type(8)))  _Float16 v8h;
typedef __attribute__((ext_vector_type(8)))  float    v8f;

#define TRES   192          // TEMPLATERES
#define REND   180          // RENDER
#define XD     90           // XDIM
#define NK     91           // XDIM+1 (rfft width)
#define KP     96           // NK padded to multiple of 16/32
#define MP     192          // 180 padded to multiple of 16/32
#define NBATCH 8
#define SCALEF (32400.0f / 7077888.0f)   // RENDER^2 / TEMPLATERES^3

// ---- workspace layout (bytes), all offsets 256B aligned ----
#define OFF_WR   0                                   // 192*192 f16 (row major)
#define OFF_WI   (OFF_WR  + MP*MP*2)
#define OFF_WN   (OFF_WI  + MP*MP*2)                 // -Wi
#define OFF_BC   (OFF_WN  + MP*MP*2)                 // 96x192, B-fragment swizzled
#define OFF_BS   (OFF_BC  + KP*MP*2)
#define OFF_XR   (OFF_BS  + KP*MP*2)                 // 8 x (192x96), B-frag swizzled
#define OFF_XI   (OFF_XR  + NBATCH*MP*KP*2)
#define OFF_YR   (OFF_XI  + NBATCH*MP*KP*2)          // 8 x (192x96), row major
#define OFF_YI   (OFF_YR  + NBATCH*MP*KP*2)
#define OFF_R    (OFF_YI  + NBATCH*MP*KP*2)          // 8*192*192 f32

// ---------------------------------------------------------------------------
// B-fragment swizzle: for a K x N matrix (K mult of 32, N mult of 16) used as
// the WMMA B operand, element (k, n) lives at:
//   kt=k/32, kk=k%32, nt=n/16, nn=n%16, lane=(kk/16)*16+nn, i=kk%16
//   idx = (((kt*NT + nt)*32) + lane)*16 + i        (NT = N/16)
// so lane L of tile (kt,nt) reads 16 contiguous halves (32 B, 32B-aligned).
// ---------------------------------------------------------------------------
__device__ __forceinline__ size_t b_swz(int k, int n, int NT) {
    int kt = k >> 5, kk = k & 31, nt = n >> 4, nn = n & 15;
    int lane = ((kk >> 4) << 4) | nn;
    return ((((size_t)kt * NT + nt) * 32) + lane) * 16 + (kk & 15);
}

// ---------------------------------------------------------------------------
// K0: build DFT weight matrices (f16, zero padded).
//   Wr[j][k] =  cos(2pi jk/180)/180   (j,k < 180)       Y = W @ X (ifft)
//   Wi[j][k] =  sin(2pi jk/180)/180,  Wn = -Wi          (row major: A operand)
//   Bc[k][n] =  w_k cos(2pi kn/180)/180  (k<91, n<180)  r = Yr@Bc + Yi@Bs
//   Bs[k][n] = -w_k sin(2pi kn/180)/180  (B-frag swizzled), w_0=w_90=1 else 2
// ---------------------------------------------------------------------------
__global__ void k_weights(_Float16* __restrict__ Wr, _Float16* __restrict__ Wi,
                          _Float16* __restrict__ Wn, _Float16* __restrict__ Bc,
                          _Float16* __restrict__ Bs) {
    const int t = blockIdx.x * blockDim.x + threadIdx.x;
    const float step = 6.28318530717958647692f / (float)REND;
    if (t < MP * MP) {
        int j = t / MP, k = t - j * MP;
        float wr = 0.f, wi = 0.f;
        if (j < REND && k < REND) {
            int p = (j * k) % REND;               // exact phase reduction
            float ang = (float)p * step;
            wr = cosf(ang) * (1.0f / 180.0f);
            wi = sinf(ang) * (1.0f / 180.0f);
        }
        Wr[t] = (_Float16)wr;
        Wi[t] = (_Float16)wi;
        Wn[t] = (_Float16)(-wi);
    }
    if (t < KP * MP) {
        int k = t / MP, n = t - k * MP;           // K-dim = k (96), N-dim = n (192)
        float bc = 0.f, bs = 0.f;
        if (k < NK && n < REND) {
            float w = (k == 0 || k == XD) ? 1.0f : 2.0f;
            int p = (k * n) % REND;
            float ang = (float)p * step;
            bc =  w * cosf(ang) * (1.0f / 180.0f);
            bs = -w * sinf(ang) * (1.0f / 180.0f);
        }
        size_t idx = b_swz(k, n, MP / 16);
        Bc[idx] = (_Float16)bc;
        Bs[idx] = (_Float16)bs;
    }
}

// ---------------------------------------------------------------------------
// K1: rotate slice grid, trilinear-sample src (2 ch = re/im), scale,
//     fftshift rows (index map i=(j+90)%180), multiply by conj(ref) -> cross,
//     stored f16 zero-padded directly in B-fragment layout (8 x 192 x 96;
//     K-dim = j, N-dim = k).
// ---------------------------------------------------------------------------
__global__ void k_sample(const float* __restrict__ src, const float* __restrict__ rot,
                         const float* __restrict__ ref,
                         _Float16* __restrict__ Xr, _Float16* __restrict__ Xi) {
    const int per = MP * KP;
    int t = blockIdx.x * blockDim.x + threadIdx.x;
    int b = t / per;
    if (b >= NBATCH) return;
    int rem = t - b * per;
    int j = rem / KP;
    int k = rem - j * KP;

    _Float16 cr = (_Float16)0.f, ci = (_Float16)0.f;
    if (j < REND && k < NK) {
        int i = (j + XD) % REND;                   // fftshift source row
        float xg = (float)k / 96.0f;
        float yg = (float)(i - XD) / 96.0f;        // z component of grid is 0
        const float* R = rot + b * 9;              // row-major 3x3
        float p0 = xg * R[0] + yg * R[3];
        float p1 = xg * R[1] + yg * R[4];
        float p2 = xg * R[2] + yg * R[5];
        float fx = (p0 + 1.0f) * 0.5f * 191.0f;
        float fy = (p1 + 1.0f) * 0.5f * 191.0f;
        float fz = (p2 + 1.0f) * 0.5f * 191.0f;
        float x0 = floorf(fx), y0 = floorf(fy), z0 = floorf(fz);
        float wx = fx - x0, wy = fy - y0, wz = fz - z0;

        const float* v0 = src + (size_t)b * 2u * TRES * TRES * TRES;
        const float* v1 = v0 + (size_t)TRES * TRES * TRES;
        float s0 = 0.f, s1 = 0.f;
        #pragma unroll
        for (int dz = 0; dz < 2; ++dz)
          #pragma unroll
          for (int dy = 0; dy < 2; ++dy)
            #pragma unroll
            for (int dx = 0; dx < 2; ++dx) {
                float zi = z0 + (float)dz, yi = y0 + (float)dy, xi = x0 + (float)dx;
                float w = (dz ? wz : 1.f - wz) * (dy ? wy : 1.f - wy)
                        * (dx ? wx : 1.f - wx);
                bool ok = (zi >= 0.f) && (zi <= 191.f) && (yi >= 0.f) &&
                          (yi <= 191.f) && (xi >= 0.f) && (xi <= 191.f);
                if (ok && w != 0.f) {
                    size_t off = ((size_t)(int)zi * TRES + (size_t)(int)yi) * TRES
                               + (size_t)(int)xi;
                    s0 += v0[off] * w;
                    s1 += v1[off] * w;
                }
            }
        float sr = s0 * SCALEF, si = s1 * SCALEF;
        size_t ro = ((size_t)b * REND + (size_t)j) * NK * 2 + (size_t)k * 2;
        float rr = ref[ro + 0];
        float rg = ref[ro + 1];
        cr = (_Float16)(sr * rr + si * rg);        // sH * conj(ref)
        ci = (_Float16)(si * rr - sr * rg);
    }
    size_t idx = (size_t)b * MP * KP + b_swz(j, k, KP / 16);
    Xr[idx] = cr;
    Xi[idx] = ci;
}

// ---------------------------------------------------------------------------
// WMMA fragment loaders, per CDNA5 ISA 7.12.2 (wave32).
//   A (16x32 f16): lane {lo=lane&15 -> M, hi=lane>>4}
//                  a[i] = A[M][ 16*hi + i ]  (contiguous k, same order as B swizzle)
//                  -> two aligned 16B chunks per lane (b128 loads)
//   B (32x16 f16): pre-swizzled -> one aligned 32B chunk per lane
//   C/D (16x16 f32): c[v] = C[ v + 8*hi ][ lo ]
// ---------------------------------------------------------------------------
__device__ __forceinline__ v16h load_a(const _Float16* __restrict__ M, int ld,
                                       int mb, int kb) {
    int lane = (int)(threadIdx.x & 31u);
    int lo = lane & 15, hi = lane >> 4;
    const _Float16* row = M + (size_t)(mb + lo) * ld + kb + 16 * hi;
    v8h c0 = *(const v8h*)(row);        // K = kb+16*hi .. +7  (16B aligned)
    v8h c1 = *(const v8h*)(row + 8);    // K = +8 .. +15       (16B aligned)
    v16h a;
    #pragma unroll
    for (int i = 0; i < 8; ++i) { a[i] = c0[i]; a[8 + i] = c1[i]; }
    return a;
}

__device__ __forceinline__ v16h load_b_swz(const _Float16* __restrict__ F, int NT,
                                           int kt, int nt) {
    int lane = (int)(threadIdx.x & 31u);
    const _Float16* p = F + ((((size_t)kt * NT + nt) * 32) + lane) * 16;
    return *(const v16h*)p;             // 32 B contiguous, 32B aligned
}

__device__ __forceinline__ v8f wmma_f16(v16h a, v16h b, v8f c) {
    return __builtin_amdgcn_wmma_f32_16x16x32_f16(
        /*neg_a=*/false, a, /*neg_b=*/false, b,
        /*c_mod=*/(short)0, c, /*reuse_a=*/false, /*reuse_b=*/false);
}

// ---------------------------------------------------------------------------
// K2: complex column transform (ifft over axis -2) as 4 chained WMMA GEMMs.
//   Yr = Wr@Xr + (-Wi)@Xi ;  Yi = Wr@Xi + Wi@Xr.   M=192, N=96, K=192.
// 256-thread blocks = 8 waves; one 16x16 output tile per wave.
// ---------------------------------------------------------------------------
__global__ void k_gemmA(const _Float16* __restrict__ Wr, const _Float16* __restrict__ Wi,
                        const _Float16* __restrict__ Wn, const _Float16* __restrict__ Xr,
                        const _Float16* __restrict__ Xi,
                        _Float16* __restrict__ Yr, _Float16* __restrict__ Yi) {
    int tile = blockIdx.x * 8 + (int)(threadIdx.x >> 5);
    int nt = tile % (KP / 16); tile /= (KP / 16);
    int mt = tile % (MP / 16); tile /= (MP / 16);
    int b  = tile;
    const int mb = mt * 16;
    const _Float16* xr = Xr + (size_t)b * MP * KP;
    const _Float16* xi = Xi + (size_t)b * MP * KP;

    v8f accR = {}; v8f accI = {};
    #pragma unroll
    for (int kt = 0; kt < MP / 32; ++kt) {
        const int kb = kt * 32;
        v16h aWr = load_a(Wr, MP, mb, kb);
        v16h aWi = load_a(Wi, MP, mb, kb);
        v16h aWn = load_a(Wn, MP, mb, kb);
        v16h bR  = load_b_swz(xr, KP / 16, kt, nt);
        v16h bI  = load_b_swz(xi, KP / 16, kt, nt);
        accR = wmma_f16(aWr, bR, accR);
        accR = wmma_f16(aWn, bI, accR);
        accI = wmma_f16(aWr, bI, accI);
        accI = wmma_f16(aWi, bR, accI);
    }
    int lane = (int)(threadIdx.x & 31u), lo = lane & 15, hi = lane >> 4;
    _Float16* yr = Yr + (size_t)b * MP * KP;
    _Float16* yi = Yi + (size_t)b * MP * KP;
    #pragma unroll
    for (int v = 0; v < 8; ++v) {
        int m = mb + v + 8 * hi;
        yr[(size_t)m * KP + nt * 16 + lo] = (_Float16)accR[v];
        yi[(size_t)m * KP + nt * 16 + lo] = (_Float16)accI[v];
    }
}

// ---------------------------------------------------------------------------
// K3: row transform (irfft over axis -1): r = Yr@Bc + Yi@Bs.
//   M=192, N=192, K=96. 8 waves/block, one 16x16 tile per wave; f32 result.
// ---------------------------------------------------------------------------
__global__ void k_gemmB(const _Float16* __restrict__ Yr, const _Float16* __restrict__ Yi,
                        const _Float16* __restrict__ Bc, const _Float16* __restrict__ Bs,
                        float* __restrict__ Rm) {
    int tile = blockIdx.x * 8 + (int)(threadIdx.x >> 5);
    int nt = tile % (MP / 16); tile /= (MP / 16);
    int mt = tile % (MP / 16); tile /= (MP / 16);
    int b  = tile;
    const int mb = mt * 16;
    const _Float16* yr = Yr + (size_t)b * MP * KP;
    const _Float16* yi = Yi + (size_t)b * MP * KP;

    v8f acc = {};
    #pragma unroll
    for (int kt = 0; kt < KP / 32; ++kt) {
        const int kb = kt * 32;
        v16h aR = load_a(yr, KP, mb, kb);
        v16h aI = load_a(yi, KP, mb, kb);
        v16h bC = load_b_swz(Bc, MP / 16, kt, nt);
        v16h bS = load_b_swz(Bs, MP / 16, kt, nt);
        acc = wmma_f16(aR, bC, acc);
        acc = wmma_f16(aI, bS, acc);
    }
    int lane = (int)(threadIdx.x & 31u), lo = lane & 15, hi = lane >> 4;
    float* out = Rm + (size_t)b * MP * MP;
    #pragma unroll
    for (int v = 0; v < 8; ++v) {
        int m = mb + v + 8 * hi;
        out[(size_t)m * MP + nt * 16 + lo] = acc[v];
    }
}

// ---------------------------------------------------------------------------
// K4: per-batch top-4 over the 180x180 region, JAX tie-break (lowest index).
// out[0:32] = 2*vals (8x4), out[32:96] = [ind%180-90, ind/180-90] (8x4x2).
// ---------------------------------------------------------------------------
__global__ void k_topk(const float* __restrict__ Rm, float* __restrict__ out) {
    const int b = blockIdx.x;
    const float* r = Rm + (size_t)b * MP * MP;
    float tv[4]; int ti[4];
    #pragma unroll
    for (int s = 0; s < 4; ++s) { tv[s] = -INFINITY; ti[s] = 0x7FFFFFFF; }

    for (int t = (int)threadIdx.x; t < REND * REND; t += (int)blockDim.x) {
        int j = t / REND, n = t - j * REND;
        float val = r[(size_t)j * MP + n];
        int   idx = t;
        #pragma unroll
        for (int s = 0; s < 4; ++s) {
            bool better = (val > tv[s]) || (val == tv[s] && idx < ti[s]);
            if (better) {
                float fv = tv[s]; int fi = ti[s];
                tv[s] = val; ti[s] = idx;
                val = fv; idx = fi;
            }
        }
    }

    __shared__ float sv[256 * 4];
    __shared__ int   si[256 * 4];
    const int tid = (int)threadIdx.x;
    #pragma unroll
    for (int s = 0; s < 4; ++s) { sv[tid * 4 + s] = tv[s]; si[tid * 4 + s] = ti[s]; }
    __syncthreads();

    for (int stride = 128; stride > 0; stride >>= 1) {
        if (tid < stride) {
            #pragma unroll
            for (int s = 0; s < 4; ++s) {
                float val = sv[(tid + stride) * 4 + s];
                int   idx = si[(tid + stride) * 4 + s];
                #pragma unroll
                for (int q = 0; q < 4; ++q) {
                    float cv = sv[tid * 4 + q]; int ci = si[tid * 4 + q];
                    bool better = (val > cv) || (val == cv && idx < ci);
                    if (better) {
                        sv[tid * 4 + q] = val; si[tid * 4 + q] = idx;
                        val = cv; idx = ci;
                    }
                }
            }
        }
        __syncthreads();
    }

    if (tid == 0) {
        #pragma unroll
        for (int s = 0; s < 4; ++s) {
            float val = sv[s]; int idx = si[s];
            int n = idx % REND, j = idx / REND;
            out[b * 4 + s]                = 2.0f * val;
            out[32 + (b * 4 + s) * 2 + 0] = (float)(n - XD);
            out[32 + (b * 4 + s) * 2 + 1] = (float)(j - XD);
        }
    }
}

// ---------------------------------------------------------------------------
extern "C" void kernel_launch(void* const* d_in, const int* in_sizes, int n_in,
                              void* d_out, int out_size, void* d_ws, size_t ws_size,
                              hipStream_t stream) {
    (void)in_sizes; (void)n_in; (void)out_size; (void)ws_size;
    const float* src = (const float*)d_in[0];   // (8,2,192,192,192)
    const float* rot = (const float*)d_in[1];   // (8,1,1,3,3)
    const float* ref = (const float*)d_in[2];   // (8,1,180,91,2)
    // d_in[3] = gridF: regenerated analytically on device.

    char* ws = (char*)d_ws;
    _Float16* Wr = (_Float16*)(ws + OFF_WR);
    _Float16* Wi = (_Float16*)(ws + OFF_WI);
    _Float16* Wn = (_Float16*)(ws + OFF_WN);
    _Float16* Bc = (_Float16*)(ws + OFF_BC);
    _Float16* Bs = (_Float16*)(ws + OFF_BS);
    _Float16* Xr = (_Float16*)(ws + OFF_XR);
    _Float16* Xi = (_Float16*)(ws + OFF_XI);
    _Float16* Yr = (_Float16*)(ws + OFF_YR);
    _Float16* Yi = (_Float16*)(ws + OFF_YI);
    float*    Rm = (float*)   (ws + OFF_R);
    float*    out = (float*)d_out;

    // K0: DFT weights (covers MP*MP = 36864 threads)
    k_weights<<<(MP * MP + 255) / 256, 256, 0, stream>>>(Wr, Wi, Wn, Bc, Bs);

    // K1: sample + cross  (8 * 192 * 96 points)
    k_sample<<<(NBATCH * MP * KP) / 256, 256, 0, stream>>>(src, rot, ref, Xr, Xi);

    // K2: column ifft GEMMs — 576 tiles, 8 waves per block
    k_gemmA<<<NBATCH * (MP / 16) * (KP / 16) / 8, 256, 0, stream>>>(Wr, Wi, Wn, Xr, Xi, Yr, Yi);

    // K3: row irfft GEMMs — 1152 tiles, 8 waves per block
    k_gemmB<<<NBATCH * (MP / 16) * (MP / 16) / 8, 256, 0, stream>>>(Yr, Yi, Bc, Bs, Rm);

    // K4: top-4 per batch
    k_topk<<<NBATCH, 256, 0, stream>>>(Rm, out);
}
